// CrossAttentionConv2d_52484500357264
// MI455X (gfx1250) — matrix operand order repaired and checked by reading, not compiled, not run on hardware
//
#include <hip/hip_runtime.h>
#include <hip/hip_bf16.h>

#define HW    2304
#define CIN   256
#define OC    512
#define NH    8
#define HD    64
#define BATCH 4
#define NT    (HW / 64)   // 36 tiles of 64

typedef __attribute__((ext_vector_type(16))) _Float16 v16h;
typedef __attribute__((ext_vector_type(8)))  float    v8f;

union FragA { v16h h; uint2 q[4]; };
union FragB { v16h h; uint4 q[2]; };
union H4    { _Float16 h[4]; uint2 u; };

// ---- CDNA5 async global->LDS copies (ASYNCcnt-tracked, no VGPR staging) ----
__device__ inline unsigned lds_addr32(const void* p) {
  // generic LDS pointer = {SHARED_BASE hi32, lds offset lo32}; low 32 bits are
  // the LDS byte address the async instructions consume.
  return (unsigned)(unsigned long long)p;
}

// one wave-instruction moves 32 lanes x 16B = 512B; INST_OFFSET applies to
// both the LDS and the global address, so offsets walk a 64B row chunk.
__device__ inline void async_copy_64B(unsigned lds, const void* g) {
  asm volatile(
      "global_load_async_to_lds_b128 %0, %1, off\n\t"
      "global_load_async_to_lds_b128 %0, %1, off offset:16\n\t"
      "global_load_async_to_lds_b128 %0, %1, off offset:32\n\t"
      "global_load_async_to_lds_b128 %0, %1, off offset:48"
      :: "v"(lds), "v"(g) : "memory");
}
__device__ inline void async_copy_32B(unsigned lds, const void* g) {
  asm volatile(
      "global_load_async_to_lds_b128 %0, %1, off\n\t"
      "global_load_async_to_lds_b128 %0, %1, off offset:16"
      :: "v"(lds), "v"(g) : "memory");
}
#define WAIT_ASYNC(n) asm volatile("s_wait_asynccnt %0" :: "i"(n) : "memory")

// A-matrix 16x32 f16 fragment from row-major [m][k] LDS tile (ISA 7.12.2).
__device__ inline v16h lds_load_a(const _Float16* lds, int stride, int mbase, int kbase) {
  int lane = threadIdx.x & 31;
  int g = lane >> 4;
  int m = mbase + (lane & 15);
  const _Float16* p = lds + m * stride + kbase + 16 * g;
  FragA f;
  f.q[0] = *(const uint2*)(p);
  f.q[1] = *(const uint2*)(p + 4);
  f.q[2] = *(const uint2*)(p + 8);
  f.q[3] = *(const uint2*)(p + 12);
  return f.h;
}

// B-matrix 32x16 f16 fragment from row-major [n][k] LDS tile.
__device__ inline v16h lds_load_b(const _Float16* lds, int stride, int nbase, int kbase) {
  int lane = threadIdx.x & 31;
  int g = lane >> 4;
  int n = nbase + (lane & 15);
  const _Float16* p = lds + n * stride + kbase + 16 * g;
  FragB f;
  f.q[0] = *(const uint4*)(p);
  f.q[1] = *(const uint4*)(p + 8);
  return f.h;
}

__device__ inline v8f wmma_f16(v16h a, v16h b, v8f c) {
  return __builtin_amdgcn_wmma_f32_16x16x32_f16(false, a, false, b, (short)0, c, false, false);
}

// ---------------------------------------------------------------------------
// Kernel 1: fused QKV 1x1-conv projection.
// grid (36, 8, 4) = (spatial tile, head, batch), 128 threads (4 waves).
// Q: (Wq x + bq + pos) * 0.125 -> Qh[b][h][p][c]
// K: (Wk x + bk)               -> Kh[b][h][p][c]
// V: (Wv x + bv)               -> Vh[b][h][c][p]   (B-frag orientation for PV)
// ---------------------------------------------------------------------------
__global__ __launch_bounds__(128) void qkv_proj_kernel(
    const float* __restrict__ x,
    const float* __restrict__ wq, const float* __restrict__ bq,
    const float* __restrict__ wk, const float* __restrict__ bk,
    const float* __restrict__ wv, const float* __restrict__ bv,
    const float* __restrict__ pos,
    _Float16* __restrict__ Qh, _Float16* __restrict__ Kh, _Float16* __restrict__ Vh)
{
  const int pt = blockIdx.x, h = blockIdx.y, b = blockIdx.z;
  const int p0 = pt * 64;
  const int tid = threadIdx.x;
  const int w = tid >> 5, lane = tid & 31;

  __shared__ __align__(16) _Float16 Xt[64][40];      // [p][i] (x transposed)
  __shared__ __align__(16) _Float16 Wt[3][64][40];   // [o][i]

  const float* xb = x + (size_t)b * CIN * HW;
  const float* wptr[3] = {wq, wk, wv};

  v8f acc[3][4];
#pragma unroll
  for (int s = 0; s < 3; ++s)
#pragma unroll
    for (int t = 0; t < 4; ++t) acc[s][t] = {};

  for (int kc = 0; kc < CIN / 32; ++kc) {
    const int i0 = kc * 32;
    __syncthreads();
    // Xt: 64p x 32i, float4 reads, transpose on store
#pragma unroll
    for (int e = 0; e < 4; ++e) {
      int idx = tid + e * 128;               // 512 float4 chunks
      int il = idx >> 4, pl = (idx & 15) * 4;
      float4 xv = *(const float4*)&xb[(size_t)(i0 + il) * HW + p0 + pl];
      Xt[pl + 0][il] = (_Float16)xv.x;
      Xt[pl + 1][il] = (_Float16)xv.y;
      Xt[pl + 2][il] = (_Float16)xv.z;
      Xt[pl + 3][il] = (_Float16)xv.w;
    }
    // Wt: 64o x 32i, float4 reads, packed b64 stores
#pragma unroll
    for (int s = 0; s < 3; ++s)
#pragma unroll
      for (int e = 0; e < 4; ++e) {
        int idx = tid + e * 128;
        int ol = idx >> 3, il = (idx & 7) * 4;
        float4 wv4 = *(const float4*)&wptr[s][(size_t)(h * HD + ol) * CIN + i0 + il];
        H4 hh;
        hh.h[0] = (_Float16)wv4.x; hh.h[1] = (_Float16)wv4.y;
        hh.h[2] = (_Float16)wv4.z; hh.h[3] = (_Float16)wv4.w;
        *(uint2*)&Wt[s][ol][il] = hh.u;
      }
    __syncthreads();

    v16h a = lds_load_a(&Xt[0][0], 40, w * 16, 0);
#pragma unroll
    for (int s = 0; s < 3; ++s)
#pragma unroll
      for (int t = 0; t < 4; ++t) {
        v16h bf = lds_load_b(&Wt[s][0][0], 40, t * 16, 0);
        acc[s][t] = wmma_f16(a, bf, acc[s][t]);
      }
  }

  const int g = lane >> 4, nn = lane & 15;
  const size_t bh = (size_t)b * NH + h;
#pragma unroll
  for (int t = 0; t < 4; ++t)
#pragma unroll
    for (int v = 0; v < 8; ++v) {
      int m = v + 8 * g;
      int pg = p0 + w * 16 + m;
      int oc = t * 16 + nn;
      int och = h * HD + oc;
      float qv = (acc[0][t][v] + bq[och] + pos[och]) * 0.125f;
      float kv = acc[1][t][v] + bk[och];
      float vv = acc[2][t][v] + bv[och];
      Qh[(bh * HW + pg) * HD + oc] = (_Float16)qv;
      Kh[(bh * HW + pg) * HD + oc] = (_Float16)kv;
      Vh[(bh * HD + oc) * HW + pg] = (_Float16)vv;
    }
}

// ---------------------------------------------------------------------------
// Kernel 2: flash attention per (b, head, 64-query tile). 4 waves, 16-row
// strips, online softmax. K/V tiles double-buffered via async global->LDS:
// prefetch tile kv+1 overlaps the WMMA work on tile kv.
// ---------------------------------------------------------------------------
__global__ __launch_bounds__(128) void flash_attn_kernel(
    const _Float16* __restrict__ Qh, const _Float16* __restrict__ Kh,
    const _Float16* __restrict__ Vh, _Float16* __restrict__ AO)
{
  const int qt = blockIdx.x, h = blockIdx.y, b = blockIdx.z;
  const int i0 = qt * 64;
  const int tid = threadIdx.x, w = tid >> 5, lane = tid & 31;
  const size_t bh = (size_t)b * NH + h;

  __shared__ __align__(16) _Float16 Qt[64][72];      // [i][c]
  __shared__ __align__(16) _Float16 Kt[2][64][72];   // [j][c]  (B-frag [n][k])
  __shared__ __align__(16) _Float16 Vt[2][64][72];   // [d][j]  (B-frag [n][k])
  __shared__ __align__(16) _Float16 Pt[64][72];      // wave-private P strips

  const _Float16* qb = Qh + bh * (size_t)HW * HD;
  const _Float16* kb = Kh + bh * (size_t)HW * HD;
  const _Float16* vb = Vh + bh * (size_t)HD * HW;

  const int lrow = tid >> 1;            // 0..63
  const int lhf  = (tid & 1) * 32;      // half-row (32 halfs = 64B)

  // async: Q tile + prefetch of KV tile 0 into buffer 0
  async_copy_64B(lds_addr32(&Qt[lrow][lhf]), qb + (size_t)(i0 + lrow) * HD + lhf);
  async_copy_64B(lds_addr32(&Kt[0][lrow][lhf]), kb + (size_t)lrow * HD + lhf);
  async_copy_64B(lds_addr32(&Vt[0][lrow][lhf]), vb + (size_t)lrow * HW + lhf);

  v8f oacc[4];
#pragma unroll
  for (int t = 0; t < 4; ++t) oacc[t] = {};
  float mrun[8], lrun[8];
#pragma unroll
  for (int v = 0; v < 8; ++v) { mrun[v] = -1e30f; lrun[v] = 0.f; }

  v16h aq[2];
  const int g = lane >> 4, nn = lane & 15;

  for (int kv = 0; kv < NT; ++kv) {
    const int cur = kv & 1;
    __syncthreads();   // all waves done reading buffer (cur^1) from iter kv-1
    if (kv + 1 < NT) {
      const int j0 = (kv + 1) * 64, nb = cur ^ 1;
      async_copy_64B(lds_addr32(&Kt[nb][lrow][lhf]),
                     kb + (size_t)(j0 + lrow) * HD + lhf);
      async_copy_64B(lds_addr32(&Vt[nb][lrow][lhf]),
                     vb + (size_t)lrow * HW + j0 + lhf);
      // async completes in order: <=8 outstanding => current tile is in LDS,
      // only the 8 just-issued prefetch copies remain in flight.
      WAIT_ASYNC(8);
    } else {
      WAIT_ASYNC(0);
    }
    __syncthreads();   // every wave's share of buffer cur is visible

    if (kv == 0) {
      aq[0] = lds_load_a(&Qt[0][0], 72, w * 16, 0);
      aq[1] = lds_load_a(&Qt[0][0], 72, w * 16, 32);
    }

    // S strip = Q^T K (16 x 64), scale already folded into Q
    v8f s[4];
#pragma unroll
    for (int jt = 0; jt < 4; ++jt) {
      s[jt] = {};
      s[jt] = wmma_f16(aq[0], lds_load_b(&Kt[cur][0][0], 72, jt * 16, 0),  s[jt]);
      s[jt] = wmma_f16(aq[1], lds_load_b(&Kt[cur][0][0], 72, jt * 16, 32), s[jt]);
    }

    // online softmax: a C-fragment row spans one 16-lane half
    float corr[8];
#pragma unroll
    for (int v = 0; v < 8; ++v) {
      float t = fmaxf(fmaxf(s[0][v], s[1][v]), fmaxf(s[2][v], s[3][v]));
      t = fmaxf(t, __shfl_xor(t, 1, 32));
      t = fmaxf(t, __shfl_xor(t, 2, 32));
      t = fmaxf(t, __shfl_xor(t, 4, 32));
      t = fmaxf(t, __shfl_xor(t, 8, 32));
      float mnew = fmaxf(mrun[v], t);
      corr[v] = __expf(mrun[v] - mnew);
      mrun[v] = mnew;
      float rs = 0.f;
#pragma unroll
      for (int jt = 0; jt < 4; ++jt) {
        float p = __expf(s[jt][v] - mnew);
        s[jt][v] = p;
        rs += p;
      }
      rs += __shfl_xor(rs, 1, 32);
      rs += __shfl_xor(rs, 2, 32);
      rs += __shfl_xor(rs, 4, 32);
      rs += __shfl_xor(rs, 8, 32);
      lrun[v] = lrun[v] * corr[v] + rs;
    }
#pragma unroll
    for (int t = 0; t < 4; ++t)
#pragma unroll
      for (int v = 0; v < 8; ++v) oacc[t][v] *= corr[v];

    // stage P (C-layout) into wave-private LDS strip as [m][j] for A-frags
#pragma unroll
    for (int jt = 0; jt < 4; ++jt)
#pragma unroll
      for (int v = 0; v < 8; ++v)
        Pt[w * 16 + v + 8 * g][jt * 16 + nn] = (_Float16)s[jt][v];

    // O += P * V^T
#pragma unroll
    for (int kc = 0; kc < 2; ++kc) {
      v16h ap = lds_load_a(&Pt[0][0], 72, w * 16, kc * 32);
#pragma unroll
      for (int dt = 0; dt < 4; ++dt)
        oacc[dt] = wmma_f16(ap, lds_load_b(&Vt[cur][0][0], 72, dt * 16, kc * 32),
                            oacc[dt]);
    }
  }

  // normalize and store f16 attention output as [b][p][channel]
#pragma unroll
  for (int dt = 0; dt < 4; ++dt)
#pragma unroll
    for (int v = 0; v < 8; ++v) {
      int m = v + 8 * g;
      int ig = i0 + w * 16 + m;
      float val = oacc[dt][v] / lrun[v];
      AO[((size_t)b * HW + ig) * OC + h * HD + dt * 16 + nn] = (_Float16)val;
    }
}

// ---------------------------------------------------------------------------
// Kernel 3: output 1x1 conv  y[b][o][p] = wo[o][c] . ao[b][p][c] + bo[o]
// grid (36, 8, 4) = (p tile, o tile, batch). Activation tile comes in via
// async global->LDS, overlapped with the fp32->f16 weight tile staging.
// ---------------------------------------------------------------------------
__global__ __launch_bounds__(128) void out_proj_kernel(
    const _Float16* __restrict__ AO, const float* __restrict__ wo,
    const float* __restrict__ bo, float* __restrict__ y)
{
  const int pt = blockIdx.x, ot = blockIdx.y, b = blockIdx.z;
  const int p0 = pt * 64, o0 = ot * 64;
  const int tid = threadIdx.x, w = tid >> 5, lane = tid & 31;

  __shared__ __align__(16) _Float16 At[64][40];   // [o][c]
  __shared__ __align__(16) _Float16 Bt[64][48];   // [p][c] stride 48 -> rows 16B aligned

  const int lrow = tid >> 1;           // 0..63
  const int lhf  = (tid & 1) * 16;     // half-row (16 halfs = 32B)

  v8f acc[4];
#pragma unroll
  for (int t = 0; t < 4; ++t) acc[t] = {};

  for (int kc = 0; kc < OC / 32; ++kc) {
    const int c0 = kc * 32;
    __syncthreads();
    // async copy of the f16 activation tile (overlaps weight staging below)
    async_copy_32B(lds_addr32(&Bt[lrow][lhf]),
                   AO + ((size_t)b * HW + p0 + lrow) * OC + c0 + lhf);
    // weight tile: float4 reads, convert, packed b64 stores
#pragma unroll
    for (int e = 0; e < 4; ++e) {
      int idx = tid + e * 128;
      int ol = idx >> 3, cl = (idx & 7) * 4;
      float4 wv4 = *(const float4*)&wo[(size_t)(o0 + ol) * OC + c0 + cl];
      H4 hh;
      hh.h[0] = (_Float16)wv4.x; hh.h[1] = (_Float16)wv4.y;
      hh.h[2] = (_Float16)wv4.z; hh.h[3] = (_Float16)wv4.w;
      *(uint2*)&At[ol][cl] = hh.u;
    }
    WAIT_ASYNC(0);
    __syncthreads();

    v16h a = lds_load_a(&At[0][0], 40, w * 16, 0);
#pragma unroll
    for (int t = 0; t < 4; ++t)
      acc[t] = wmma_f16(a, lds_load_b(&Bt[0][0], 48, t * 16, 0), acc[t]);
  }

  const int g = lane >> 4, nn = lane & 15;
#pragma unroll
  for (int t = 0; t < 4; ++t)
#pragma unroll
    for (int v = 0; v < 8; ++v) {
      int m = v + 8 * g;
      int og = o0 + w * 16 + m;
      int pg = p0 + t * 16 + nn;
      y[((size_t)b * OC + og) * HW + pg] = acc[t][v] + bo[og];
    }
}

// ---------------------------------------------------------------------------
extern "C" void kernel_launch(void* const* d_in, const int* in_sizes, int n_in,
                              void* d_out, int out_size, void* d_ws, size_t ws_size,
                              hipStream_t stream) {
  const float* x   = (const float*)d_in[0];
  const float* wq  = (const float*)d_in[1];
  const float* bq  = (const float*)d_in[2];
  const float* wk  = (const float*)d_in[3];
  const float* bk  = (const float*)d_in[4];
  const float* wv  = (const float*)d_in[5];
  const float* bv  = (const float*)d_in[6];
  const float* pos = (const float*)d_in[7];
  const float* wo  = (const float*)d_in[8];
  const float* bo  = (const float*)d_in[9];
  float* out = (float*)d_out;

  const size_t QKV = (size_t)BATCH * NH * HW * HD;
  _Float16* Qh = (_Float16*)d_ws;
  _Float16* Kh = Qh + QKV;
  _Float16* Vh = Kh + QKV;
  _Float16* AO = Vh + QKV;   // [B][HW][OC] f16

  dim3 blk(128);
  qkv_proj_kernel<<<dim3(HW / 64, NH, BATCH), blk, 0, stream>>>(
      x, wq, bq, wk, bk, wv, bv, pos, Qh, Kh, Vh);
  flash_attn_kernel<<<dim3(HW / 64, NH, BATCH), blk, 0, stream>>>(Qh, Kh, Vh, AO);
  out_proj_kernel<<<dim3(HW / 64, OC / 64, BATCH), blk, 0, stream>>>(AO, wo, bo, out);
}
